// TransformerLayer_24318104830545
// MI455X (gfx1250) — matrix operand order repaired and checked by reading, not compiled, not run on hardware
//
#include <hip/hip_runtime.h>
#include <hip/hip_bf16.h>
#include <math.h>

// ---------------------------------------------------------------------------
// Problem constants
// ---------------------------------------------------------------------------
#define BB    16
#define FF    4096
#define TT    16
#define DIM   512
#define HH    8
#define HEAD  64
#define CDIM  128
#define GEQ   0.04419417382415922f   /* 1/sqrt(512) equalized-lr gain */
#define SCALE 0.125f                  /* 1/sqrt(HEAD) */

typedef __bf16 bf16_t;
typedef __attribute__((ext_vector_type(16))) __bf16 bf16x16;
typedef __attribute__((ext_vector_type(8)))  float  f32x8;
typedef __attribute__((ext_vector_type(2)))  float  f32x2;
typedef __attribute__((ext_vector_type(4)))  unsigned u32x4;
typedef __attribute__((ext_vector_type(8)))  int      i32x8;
typedef __attribute__((ext_vector_type(4)))  int      i32x4;

#if __has_builtin(__builtin_amdgcn_tensor_load_to_lds)
#define HAVE_TDM 1
#else
#define HAVE_TDM 0
#endif

union U16B { uint4 u[2]; bf16x16 v; };

// B fragment (bf16, 32x16): lane holds column n = lane&15, 16 contiguous K
// halves starting at K = (lane>>4)*16  (same k order as the A fragment).
__device__ __forceinline__ bf16x16 load_b_frag(const bf16_t* p) {
    U16B u;
    u.u[0] = ((const uint4*)p)[0];
    u.u[1] = ((const uint4*)p)[1];
    return u.v;
}
// A fragment (bf16, 16x32): lane holds row m = lane&15; halves 0..15 are
// K = hi*16..hi*16+15 (hi = lane>>4).
__device__ __forceinline__ bf16x16 load_a_frag_bf(const bf16_t* rowp, int hi) {
    U16B u;
    u.u[0] = *(const uint4*)(rowp + hi * 16);
    u.u[1] = *(const uint4*)(rowp + hi * 16 + 8);
    return u.v;
}
// Same A fragment but converting fp32 memory -> bf16 registers on the fly.
__device__ __forceinline__ bf16x16 load_a_frag_f32(const float* rowp, int hi) {
    const float4* p0 = (const float4*)(rowp + hi * 16);
    const float4* p1 = (const float4*)(rowp + hi * 16 + 8);
    float4 a = p0[0], b = p0[1], c = p1[0], d = p1[1];
    bf16x16 r;
    r[0] = (__bf16)a.x;  r[1] = (__bf16)a.y;  r[2] = (__bf16)a.z;  r[3] = (__bf16)a.w;
    r[4] = (__bf16)b.x;  r[5] = (__bf16)b.y;  r[6] = (__bf16)b.z;  r[7] = (__bf16)b.w;
    r[8] = (__bf16)c.x;  r[9] = (__bf16)c.y;  r[10] = (__bf16)c.z; r[11] = (__bf16)c.w;
    r[12] = (__bf16)d.x; r[13] = (__bf16)d.y; r[14] = (__bf16)d.z; r[15] = (__bf16)d.w;
    return r;
}
__device__ __forceinline__ f32x8 wmma_bf16(bf16x16 a, bf16x16 b, f32x8 c) {
    return __builtin_amdgcn_wmma_f32_16x16x32_bf16(false, a, false, b, (short)0, c, false, false);
}

// ---------------------------------------------------------------------------
// Elementwise: bf16 weight conversion with folded equalized-lr gain
// ---------------------------------------------------------------------------
__global__ void k_cvt_w(const float* __restrict__ in, bf16_t* __restrict__ out,
                        float scale, int n) {
    int i = blockIdx.x * blockDim.x + threadIdx.x;
    if (i < n) out[i] = (__bf16)(in[i] * scale);
}

// ---------------------------------------------------------------------------
// Generic WMMA GEMM:  C[M,N] = A[M,K] @ Wbf[N,K]^T + bias[N]  (+ epilogue)
// block = 256 threads = 8 waves; wave computes 16M x 64N; grid (M/64, N/128).
// ABF16: A operand already bf16 in memory.  EPI==1: fp32 out with residual.
// TDM:  stage the block's 128x32 W tile into LDS via the Tensor Data Mover,
//       read B fragments from LDS (contiguous ds_load_b128 pairs) so the
//       vector-memory pipe is dedicated to streaming A.
// ---------------------------------------------------------------------------
template<bool ABF16, int EPI, bool TDM>
__global__ void k_gemm(const void* __restrict__ Aptr, const bf16_t* __restrict__ W,
                       const float* __restrict__ bias, void* __restrict__ Cptr,
                       const float* __restrict__ resid, const float* __restrict__ rowmean,
                       int M, int N, int K) {
    int w = threadIdx.x >> 5, lane = threadIdx.x & 31;
    int ln = lane & 15, hi = lane >> 4;
    int m0 = blockIdx.x * 64 + (w & 3) * 16;
    int nblk = blockIdx.y * 128;
    int n0 = nblk + (w >> 2) * 64;
    int row = m0 + ln;

    f32x8 acc[4] = {};

    if constexpr (TDM && HAVE_TDM) {
        __shared__ bf16_t wtile[128 * 32];     // [n_local][k_local] row-major, 8KB
        unsigned ldsaddr = (unsigned)(size_t)(void*)wtile;
        int nloc = (w >> 2) * 64;
        for (int kk = 0; kk < K; kk += 32) {
            __syncthreads();                   // previous tile fully consumed
            if (w == 0) {
                unsigned long long ga =
                    (unsigned long long)(size_t)(const void*)(W + (size_t)nblk * K + kk);
                // D# group0: count=1, lds_addr, 57-bit global addr, type=2
                u32x4 g0;
                g0[0] = 1u;
                g0[1] = ldsaddr;
                g0[2] = (unsigned)ga;
                g0[3] = ((unsigned)(ga >> 32) & 0x01FFFFFFu) | 0x80000000u;
                // D# group1: data_size=2B; tensor_dim0=512, tensor_dim1=1M (no OOB);
                // tile = 32 (k) x 128 (n rows); row stride = K = 512 elements
                i32x8 g1;
                g1[0] = (int)(1u << 16);            // data_size code 1 = 2 bytes
                g1[1] = (int)(512u << 16);          // tensor_dim0 = 512 (bits 79:48)
                g1[2] = 0;                          // tensor_dim1 low16 = 0
                g1[3] = (int)(16u | (32u << 16));   // tensor_dim1 hi = 16 (=1M), tile_dim0 = 32
                g1[4] = 128;                        // tile_dim1 = 128, tile_dim2 = 0
                g1[5] = 512;                        // tensor_dim0_stride = 512 elements
                g1[6] = 0;
                g1[7] = 0;
                i32x4 gz4 = {0, 0, 0, 0};
                i32x8 gz8 = {0, 0, 0, 0, 0, 0, 0, 0};
                __builtin_amdgcn_tensor_load_to_lds(g0, g1, gz4, gz4, gz8, 0);
                __builtin_amdgcn_s_wait_tensorcnt(0);
            }
            __syncthreads();                   // tile visible to all waves

            bf16x16 af;
            if (ABF16) {
                const bf16_t* ap = (const bf16_t*)Aptr + (size_t)row * K + kk;
                __builtin_prefetch(ap + 64, 0, 1);
                af = load_a_frag_bf(ap, hi);
            } else {
                const float* ap = (const float*)Aptr + (size_t)row * K + kk;
                __builtin_prefetch(ap + 64, 0, 1);
                af = load_a_frag_f32(ap, hi);
            }
#pragma unroll
            for (int j = 0; j < 4; j++) {
                const bf16_t* wp = &wtile[(size_t)(nloc + j * 16 + ln) * 32 + hi * 16];
                acc[j] = wmma_bf16(af, load_b_frag(wp), acc[j]);
            }
        }
    } else {
        for (int kk = 0; kk < K; kk += 32) {
            bf16x16 af;
            if (ABF16) af = load_a_frag_bf((const bf16_t*)Aptr + (size_t)row * K + kk, hi);
            else       af = load_a_frag_f32((const float*)Aptr + (size_t)row * K + kk, hi);
#pragma unroll
            for (int j = 0; j < 4; j++) {
                const bf16_t* wp = W + (size_t)(n0 + j * 16 + ln) * K + kk + hi * 16;
                acc[j] = wmma_bf16(af, load_b_frag(wp), acc[j]);
            }
        }
    }

#pragma unroll
    for (int j = 0; j < 4; j++) {
        int n = n0 + j * 16 + ln;
        float bb = bias ? bias[n] : 0.0f;
#pragma unroll
        for (int r = 0; r < 8; r++) {
            size_t m = (size_t)m0 + r + 8 * hi;
            float val = acc[j][r] + bb;
            if (EPI == 0) ((bf16_t*)Cptr)[m * N + n] = (__bf16)val;
            else          ((float*)Cptr)[m * N + n] = val + resid[m * N + n] - rowmean[m];
        }
    }
}

// ---------------------------------------------------------------------------
// to-side (T=16 per batch): values -> v[B,H,T,HEAD] layout, and g_to gates
// ---------------------------------------------------------------------------
__global__ void k_to_side(const float* __restrict__ x_to, const float* __restrict__ to_pos,
                          const float* __restrict__ Wv, const float* __restrict__ bv,
                          const float* __restrict__ Wg_to, const float* __restrict__ bg_to,
                          const float* __restrict__ Wgp_to, const float* __restrict__ bgp_to,
                          float* __restrict__ v_out, float* __restrict__ g_to_out) {
    int bt = blockIdx.x;               // b*16 + t
    int b = bt >> 4, t = bt & 15;
    __shared__ float xr[DIM], pr[DIM];
    const float* xrow = x_to + (size_t)bt * DIM;
    const float* prow = to_pos + (size_t)t * DIM;
    for (int i = threadIdx.x; i < DIM; i += 256) { xr[i] = xrow[i]; pr[i] = prow[i]; }
    __syncthreads();
    for (int o = threadIdx.x; o < DIM; o += 256) {
        const float* wr = Wv + (size_t)o * DIM;
        float acc = 0.f;
        for (int k = 0; k < DIM; k++) acc += xr[k] * wr[k];
        acc = acc * GEQ + bv[o];
        int h = o >> 6, d = o & 63;
        v_out[(((size_t)(b * HH + h)) * TT + t) * HEAD + d] = acc;
    }
    if (threadIdx.x < HH) {
        int h = threadIdx.x;
        const float* w1 = Wg_to + (size_t)h * DIM;
        const float* w2 = Wgp_to + (size_t)h * DIM;
        float a1 = 0.f, a2 = 0.f;
        for (int k = 0; k < DIM; k++) { a1 += xr[k] * w1[k]; a2 += pr[k] * w2[k]; }
        float z = a1 * GEQ + bg_to[h] + a2 * GEQ + bgp_to[h]; // GATE_BIAS_TO = 0
        g_to_out[(b * HH + h) * TT + t] = 1.f / (1.f + __expf(-z));
    }
}

// gposfr[f][h] = fc(from_pos, Wgp_fr)  -- batch independent
__global__ void k_gposfr(const float* __restrict__ from_pos, const float* __restrict__ Wgp_fr,
                         const float* __restrict__ bgp_fr, float* __restrict__ gposfr) {
    int idx = blockIdx.x * blockDim.x + threadIdx.x;   // FF*HH
    if (idx >= FF * HH) return;
    int f = idx >> 3, h = idx & 7;
    const float* xr = from_pos + (size_t)f * DIM;
    const float* wr = Wgp_fr + (size_t)h * DIM;
    float acc = 0.f;
    for (int k = 0; k < DIM; k++) acc += xr[k] * wr[k];
    gposfr[idx] = acc * GEQ + bgp_fr[h];
}

// ---------------------------------------------------------------------------
// One pass over x_from per row: g_fr gates (8 dots) + row mean. Wave per row.
// ---------------------------------------------------------------------------
__global__ void k_row_pass(const float* __restrict__ x_from, const float* __restrict__ Wg_fr,
                           const float* __restrict__ bg_fr, const float* __restrict__ gposfr,
                           float* __restrict__ g_fr, float* __restrict__ rowmean) {
    __shared__ float wl[HH * DIM];
    for (int i = threadIdx.x; i < HH * DIM; i += 256) wl[i] = Wg_fr[i];
    __syncthreads();
    int w = threadIdx.x >> 5, lane = threadIdx.x & 31;
    size_t m = (size_t)blockIdx.x * 8 + w;
    const float* xr = x_from + m * DIM;
    float a[9] = {};
    for (int k = lane; k < DIM; k += 32) {
        float xv = xr[k];
        a[8] += xv;
#pragma unroll
        for (int h = 0; h < HH; h++) a[h] += xv * wl[h * DIM + k];
    }
#pragma unroll
    for (int h = 0; h < 9; h++)
#pragma unroll
        for (int off = 16; off > 0; off >>= 1) a[h] += __shfl_xor(a[h], off);
    if (lane == 0) rowmean[m] = a[8] * (1.0f / DIM);
    if (lane < HH) {
        int h = lane, fl = (int)(m & (FF - 1));
        float z = a[h] * GEQ + bg_fr[h] + gposfr[fl * HH + h] + 1.0f; // GATE_BIAS_FROM
        g_fr[m * HH + h] = 1.f / (1.f + __expf(-z));
    }
}

// wcent0[h][t][c] = att_weight[h][c] * centroids[h][t][c]  (bf16)
__global__ void k_wcent0(const float* __restrict__ aw, const float* __restrict__ cen,
                         bf16_t* __restrict__ out) {
    int idx = blockIdx.x * blockDim.x + threadIdx.x;   // HH*TT*CDIM
    if (idx >= HH * TT * CDIM) return;
    int c = idx & (CDIM - 1), h = idx >> 11;
    out[idx] = (__bf16)(aw[h * CDIM + c] * cen[idx]);
}
// wcent1[bh][t][c] = att_weight[h][c] * to_centroids[bh][t][c]  (bf16)
__global__ void k_wcent1(const float* __restrict__ aw, const float* __restrict__ tc,
                         bf16_t* __restrict__ out) {
    int idx = blockIdx.x * blockDim.x + threadIdx.x;   // BB*HH*TT*CDIM
    if (idx >= BB * HH * TT * CDIM) return;
    int c = idx & (CDIM - 1), h = (idx >> 11) & 7;
    out[idx] = (__bf16)(aw[h * CDIM + c] * tc[idx]);
}

// ---------------------------------------------------------------------------
// Scores + softmax (+ gating on ITER==1). Wave per 16 from-rows per (b,h).
// from_elements[b,h,f,c] = h<4 ? _q_bf[b*F+f][h*128+c] : qpos_bf[f][(h-4)*128+c]
// ---------------------------------------------------------------------------
template<int ITER>
__global__ void k_scores(const bf16_t* __restrict__ qbf, const bf16_t* __restrict__ qposbf,
                         const bf16_t* __restrict__ wcent, const float* __restrict__ g_to,
                         const float* __restrict__ g_fr, float* __restrict__ probs) {
    int bh = blockIdx.x, b = bh >> 3, h = bh & 7;
    int w = threadIdx.x >> 5, lane = threadIdx.x & 31;
    int ln = lane & 15, hi = lane >> 4;
    int f0 = blockIdx.y * 128 + w * 16;
    int frow = f0 + ln;
    size_t m = (size_t)b * FF + frow;
    const bf16_t* arow = (h < 4) ? (qbf + m * DIM + h * CDIM)
                                 : (qposbf + (size_t)frow * DIM + (h - 4) * CDIM);
    const bf16_t* wb = wcent + (((ITER ? (size_t)bh : (size_t)h) * TT) + ln) * CDIM;

    f32x8 acc = {};
    for (int kk = 0; kk < CDIM; kk += 32)
        acc = wmma_bf16(load_a_frag_bf(arow + kk, hi), load_b_frag(wb + kk + hi * 16), acc);

    __shared__ float lds[8][16][17];
#pragma unroll
    for (int r = 0; r < 8; r++) lds[w][r + 8 * hi][ln] = acc[r];
    __syncthreads();

    if (lane < 16) {
        int fr = f0 + lane;
        float vals[16], mx = -1e30f;
#pragma unroll
        for (int t = 0; t < TT; t++) { vals[t] = lds[w][lane][t] * SCALE; mx = fmaxf(mx, vals[t]); }
        float s = 0.f;
#pragma unroll
        for (int t = 0; t < TT; t++) { vals[t] = __expf(vals[t] - mx); s += vals[t]; }
        float inv = 1.f / s;
        float gf = 1.f;
        if (ITER) gf = g_fr[((size_t)b * FF + fr) * HH + h];
        float* op = probs + ((size_t)bh * FF + fr) * TT;
#pragma unroll
        for (int t = 0; t < TT; t++) {
            float p = vals[t] * inv;
            if (ITER) p *= gf * g_to[bh * TT + t];
            op[t] = p;
        }
    }
}

// inv[bh][t] = 1 / (sum_f probs[bh][f][t] + 1e-8)
__global__ void k_denominv(const float* __restrict__ probs, float* __restrict__ inv) {
    int bh = blockIdx.x;
    int t = threadIdx.x & 15, part = threadIdx.x >> 4;
    const float* p = probs + (size_t)bh * FF * TT;
    float s = 0.f;
    for (int f = part; f < FF; f += 16) s += p[(size_t)f * TT + t];
    __shared__ float lds[256];
    lds[threadIdx.x] = s;
    __syncthreads();
    if (threadIdx.x < 16) {
        float tot = 0.f;
        for (int i = 0; i < 16; i++) tot += lds[i * 16 + threadIdx.x];
        inv[bh * TT + threadIdx.x] = 1.f / (tot + 1e-8f);
    }
}

// ---------------------------------------------------------------------------
// Centroid update: tc[bh][t][c] += sum_f assign[t][f] * fe[f][c]; K split 16x.
// One wave per block; f32 atomics into tc.
// ---------------------------------------------------------------------------
__global__ void k_centroid_update(const float* __restrict__ probs, const float* __restrict__ inv0,
                                  const bf16_t* __restrict__ qbf, const bf16_t* __restrict__ qposbf,
                                  float* __restrict__ tc) {
    int bh = blockIdx.x, b = bh >> 3, h = bh & 7;
    int ks = blockIdx.y;                       // f chunk [ks*256, ks*256+256)
    int lane = threadIdx.x & 31, ln = lane & 15, hi = lane >> 4;
    float sc = inv0[bh * TT + ln];             // A row t = ln
    const float* pbase = probs + (size_t)bh * FF * TT;
    const bf16_t* febase = (h < 4) ? (qbf + (size_t)b * FF * DIM + h * CDIM)
                                   : (qposbf + (size_t)(h - 4) * CDIM);
    f32x8 acc[8] = {};
    for (int kk = ks * 256; kk < ks * 256 + 256; kk += 32) {
        bf16x16 af;
#pragma unroll
        for (int jj = 0; jj < 16; jj++) {
            int f = kk + hi * 16 + jj;
            af[jj] = (__bf16)(pbase[(size_t)f * TT + ln] * sc);
        }
#pragma unroll
        for (int j = 0; j < 8; j++) {
            int c = j * 16 + ln;
            const bf16_t* fe = febase + c;
            bf16x16 bfr;
#pragma unroll
            for (int jj = 0; jj < 16; jj++) bfr[jj] = fe[(size_t)(kk + hi * 16 + jj) * DIM];
            acc[j] = wmma_bf16(af, bfr, acc[j]);
        }
    }
#pragma unroll
    for (int j = 0; j < 8; j++) {
        int c = j * 16 + ln;
#pragma unroll
        for (int r = 0; r < 8; r++) {
            int t = r + 8 * hi;
            atomicAdd(&tc[((size_t)bh * TT + t) * CDIM + c], acc[j][r]);
        }
    }
}

// out2[b][f][t] = mean_h gated_probs[b][h][f][t]
__global__ void k_out2(const float* __restrict__ probs, float* __restrict__ out2) {
    int idx = blockIdx.x * blockDim.x + threadIdx.x;     // BB*FF*TT
    if (idx >= BB * FF * TT) return;
    int t = idx & 15, f = (idx >> 4) & (FF - 1), b = idx >> 16;
    float s = 0.f;
#pragma unroll
    for (int h = 0; h < HH; h++) s += probs[(((size_t)(b * HH + h)) * FF + f) * TT + t];
    out2[idx] = s * 0.125f;
}

// out3[bh][t][f] = gated_probs[bh][f][t] * inv1[bh][t]
__global__ void k_tofrom(const float* __restrict__ probs, const float* __restrict__ inv1,
                         float* __restrict__ out3) {
    int idx = blockIdx.x * blockDim.x + threadIdx.x;     // BB*HH*TT*FF
    if (idx >= BB * HH * TT * FF) return;
    int f = idx & (FF - 1), t = (idx >> 12) & 15, bh = idx >> 16;
    out3[idx] = probs[((size_t)bh * FF + f) * TT + t] * inv1[bh * TT + t];
}

// ---------------------------------------------------------------------------
// control[b*F+f][h*64+d] = sum_t gated_probs[b,h,f,t] * v[b,h,t,d]  (bf16 out)
// Wave per 16 from-rows; K=16 via fp32 16x16x4 WMMA (exact), bf16 fallback.
// ---------------------------------------------------------------------------
__global__ void k_control(const float* __restrict__ probs, const float* __restrict__ v,
                          bf16_t* __restrict__ outBf) {
    int b = blockIdx.x;
    int w = threadIdx.x >> 5, lane = threadIdx.x & 31;
    int ln = lane & 15, hi = lane >> 4;
    int f0 = blockIdx.y * 128 + w * 16;
    for (int h = 0; h < HH; h++) {
        const float* p = probs + (((size_t)(b * HH + h)) * FF + f0) * TT;
        const float* vb = v + ((size_t)(b * HH + h)) * TT * HEAD;
        f32x8 acc[4] = {};
#if __has_builtin(__builtin_amdgcn_wmma_f32_16x16x4_f32)
#pragma unroll
        for (int kb = 0; kb < TT; kb += 4) {
            int k0 = kb + hi * 2;
            f32x2 af;
            af.x = p[(size_t)ln * TT + k0];
            af.y = p[(size_t)ln * TT + k0 + 1];
#pragma unroll
            for (int j = 0; j < 4; j++) {
                int d = j * 16 + ln;
                f32x2 bfv;
                bfv.x = vb[(size_t)k0 * HEAD + d];
                bfv.y = vb[(size_t)(k0 + 1) * HEAD + d];
                acc[j] = __builtin_amdgcn_wmma_f32_16x16x4_f32(
                             false, af, false, bfv, (short)0, acc[j], false, false);
            }
        }
#else
        // bf16 fallback: single K=32 step, zero-padded above K=16
        bf16x16 af;
#pragma unroll
        for (int jj = 0; jj < 16; jj++) {
            int k = hi * 8 + ((jj >> 3) << 4) + (jj & 7);
            af[jj] = (k < TT) ? (__bf16)p[(size_t)ln * TT + k] : (__bf16)0.f;
        }
#pragma unroll
        for (int j = 0; j < 4; j++) {
            int d = j * 16 + ln;
            bf16x16 bfr;
#pragma unroll
            for (int jj = 0; jj < 16; jj++) {
                int k = hi * 16 + jj;
                bfr[jj] = (k < TT) ? (__bf16)vb[(size_t)k * HEAD + d] : (__bf16)0.f;
            }
            acc[j] = wmma_bf16(af, bfr, acc[j]);
        }
#endif
#pragma unroll
        for (int j = 0; j < 4; j++)
#pragma unroll
            for (int r = 0; r < 8; r++) {
                size_t m = (size_t)b * FF + f0 + r + 8 * hi;
                outBf[m * DIM + h * HEAD + j * 16 + ln] = (__bf16)acc[j][r];
            }
    }
}

// ---------------------------------------------------------------------------
// Host orchestration
// ---------------------------------------------------------------------------
extern "C" void kernel_launch(void* const* d_in, const int* in_sizes, int n_in,
                              void* d_out, int out_size, void* d_ws, size_t ws_size,
                              hipStream_t stream) {
    (void)in_sizes; (void)n_in; (void)out_size; (void)ws_size;
    const float* from_tensor = (const float*)d_in[0];
    const float* to_tensor   = (const float*)d_in[1];
    const float* from_pos    = (const float*)d_in[2];
    const float* to_pos      = (const float*)d_in[3];
    const float* Wq   = (const float*)d_in[4];  const float* bq    = (const float*)d_in[5];
    const float* Wv   = (const float*)d_in[8];  const float* bv    = (const float*)d_in[9];
    const float* Wfp  = (const float*)d_in[10]; const float* bfp   = (const float*)d_in[11];
    const float* Wg_to  = (const float*)d_in[14]; const float* bg_to  = (const float*)d_in[15];
    const float* Wgp_to = (const float*)d_in[16]; const float* bgp_to = (const float*)d_in[17];
    const float* Wg_fr  = (const float*)d_in[18]; const float* bg_fr  = (const float*)d_in[19];
    const float* Wgp_fr = (const float*)d_in[20]; const float* bgp_fr = (const float*)d_in[21];
    const float* Wmod = (const float*)d_in[22]; const float* bmod  = (const float*)d_in[23];
    const float* att_weight = (const float*)d_in[24];
    const float* centroids  = (const float*)d_in[25];

    float* out1 = (float*)d_out;                                  // [B,F,DIM]
    float* out2 = out1 + (size_t)BB * FF * DIM;                   // [B,F,T]
    float* out3 = out2 + (size_t)BB * FF * TT;                    // [B,H,T,F]

    // ---- workspace layout --------------------------------------------------
    char* base = (char*)d_ws;
    size_t off = 0;
    auto alloc = [&](size_t bytes) { void* p = base + off; off = (off + bytes + 255) & ~(size_t)255; return p; };
    bf16_t* qbf      = (bf16_t*)alloc((size_t)BB * FF * DIM * 2);   // _queries bf16; later aliased by control
    float*  probs    = (float*) alloc((size_t)BB * HH * FF * TT * 4);
    bf16_t* qposbf   = (bf16_t*)alloc((size_t)FF * DIM * 2);
    bf16_t* WqBf     = (bf16_t*)alloc((size_t)DIM * DIM * 2);
    bf16_t* WfpBf    = (bf16_t*)alloc((size_t)DIM * DIM * 2);
    bf16_t* WmodBf   = (bf16_t*)alloc((size_t)DIM * DIM * 2);
    float*  vbuf     = (float*) alloc((size_t)BB * HH * TT * HEAD * 4);
    float*  g_to     = (float*) alloc((size_t)BB * HH * TT * 4);
    float*  g_fr     = (float*) alloc((size_t)BB * FF * HH * 4);
    float*  gposfr   = (float*) alloc((size_t)FF * HH * 4);
    bf16_t* wcent0   = (bf16_t*)alloc((size_t)HH * TT * CDIM * 2);
    float*  tc       = (float*) alloc((size_t)BB * HH * TT * CDIM * 4);
    bf16_t* wcent1   = (bf16_t*)alloc((size_t)BB * HH * TT * CDIM * 2);
    float*  inv0     = (float*) alloc((size_t)BB * HH * TT * 4);
    float*  inv1     = (float*) alloc((size_t)BB * HH * TT * 4);
    float*  rowmean  = (float*) alloc((size_t)BB * FF * 4);
    bf16_t* controlBf = qbf;   // alias: _queries dead after second k_scores

    const int NW = DIM * DIM;  // 262144

    // 1. weights -> bf16 with folded 1/sqrt(fan_in)
    k_cvt_w<<<NW / 256, 256, 0, stream>>>(Wq,   WqBf,   GEQ, NW);
    k_cvt_w<<<NW / 256, 256, 0, stream>>>(Wfp,  WfpBf,  GEQ, NW);
    k_cvt_w<<<NW / 256, 256, 0, stream>>>(Wmod, WmodBf, GEQ, NW);

    // 2. to-side: values->v layout, g_to gates
    k_to_side<<<BB * TT, 256, 0, stream>>>(to_tensor, to_pos, Wv, bv,
                                           Wg_to, bg_to, Wgp_to, bgp_to, vbuf, g_to);
    // 3. batch-independent from-pos gate projection
    k_gposfr<<<(FF * HH) / 256, 256, 0, stream>>>(from_pos, Wgp_fr, bgp_fr, gposfr);

    // 4. qpos = fc(from_pos, Wfp)+bfp (batch independent!)  [4096x512] bf16
    k_gemm<false, 0, false><<<dim3(FF / 64, DIM / 128), 256, 0, stream>>>(
        from_pos, WfpBf, bfp, qposbf, nullptr, nullptr, FF, DIM, DIM);
    // 5. _queries = fc(x_from, Wq)+bq  [65536x512] bf16  (TDM-staged W tiles)
    k_gemm<false, 0, true><<<dim3(BB * FF / 64, DIM / 128), 256, 0, stream>>>(
        from_tensor, WqBf, bq, qbf, nullptr, nullptr, BB * FF, DIM, DIM);

    // 6. per-row pass: g_fr gates + row means (one sweep of x_from)
    k_row_pass<<<(BB * FF) / 8, 256, 0, stream>>>(from_tensor, Wg_fr, bg_fr,
                                                  gposfr, g_fr, rowmean);

    // 7. k-means iteration 0
    k_wcent0<<<(HH * TT * CDIM) / 256, 256, 0, stream>>>(att_weight, centroids, wcent0);
    k_scores<0><<<dim3(BB * HH, FF / 128), 256, 0, stream>>>(qbf, qposbf, wcent0,
                                                             nullptr, nullptr, probs);
    k_denominv<<<BB * HH, 256, 0, stream>>>(probs, inv0);

    // 8. centroid update (K=4096 WMMA GEMM, 16-way K-split, f32 atomics)
    (void)hipMemsetAsync(tc, 0, (size_t)BB * HH * TT * CDIM * 4, stream);
    k_centroid_update<<<dim3(BB * HH, 16), 32, 0, stream>>>(probs, inv0, qbf, qposbf, tc);
    k_wcent1<<<(BB * HH * TT * CDIM) / 256, 256, 0, stream>>>(att_weight, tc, wcent1);

    // 9. k-means iteration 1 (fused gating) -> overwrites probs with gated probs
    k_scores<1><<<dim3(BB * HH, FF / 128), 256, 0, stream>>>(qbf, qposbf, wcent1,
                                                             g_to, g_fr, probs);

    // 10. outputs 2 & 3
    k_denominv<<<BB * HH, 256, 0, stream>>>(probs, inv1);
    k_out2<<<(BB * FF * TT) / 256, 256, 0, stream>>>(probs, out2);
    k_tofrom<<<(BB * HH * TT * FF) / 256, 256, 0, stream>>>(probs, inv1, out3);

    // 11. control = gated_probs @ v  (fp32 16x16x4 WMMA) -> bf16 (aliases qbf)
    k_control<<<dim3(BB, FF / 128), 256, 0, stream>>>(probs, vbuf, controlBf);

    // 12. out = (x_from - rowmean) + fc(control, Wmod)+bmod  (TDM-staged W tiles)
    k_gemm<true, 1, true><<<dim3(BB * FF / 64, DIM / 128), 256, 0, stream>>>(
        controlBf, WmodBf, bmod, out1, from_tensor, rowmean, BB * FF, DIM, DIM);
}
